// DeformConv2d_1726576855660
// MI455X (gfx1250) — compile-verified, matrix operand order checked by reading, not run on hardware
//
#include <hip/hip_runtime.h>

typedef __attribute__((ext_vector_type(16))) __bf16 v16bf;
typedef __attribute__((ext_vector_type(8)))  __bf16 v8bf;
typedef __attribute__((ext_vector_type(8)))  float  v8f;

#define B_    4
#define C_    256
#define H_    64
#define W_    64
#define CO_   256
#define KTAPS 9
#define HW_   4096          // H*W
#define KTOT  2304          // C_*KTAPS
#define KSTEPS 72           // KTOT/32
#define NTILE 64            // pixels per workgroup
#define PIX   (B_*HW_)      // 16384 output pixels

// -------------------------------------------------------------------------
// Kernel 1: pack fp32 weights -> bf16 in WMMA A-fragment order.
// Layout: wbf[kt][mt][lane][e], kt=0..71, mt=0..15 (M=mt*16), lane=0..31, e=0..15
//   m  = mt*16 + (lane & 15)
//   kk = (e&7) + ((e>>3)<<4) + ((lane>=16)<<3)   (ISA 16-bit A 16x32 layout)
//   f  = kt*32 + kk ;  weight element = w[m*2304 + f]
// -------------------------------------------------------------------------
__global__ __launch_bounds__(256)
void pack_weights_kernel(const float* __restrict__ w, __bf16* __restrict__ wbf) {
    int idx  = blockIdx.x * 256 + threadIdx.x;      // < 72*16*32*16 = 589824
    int e    = idx & 15;
    int lane = (idx >> 4) & 31;
    int mt   = (idx >> 9) & 15;
    int kt   = idx >> 13;
    int m  = mt * 16 + (lane & 15);
    int kk = (e & 7) + ((e >> 3) << 4) + ((lane >> 4) << 3);
    int f  = kt * 32 + kk;
    wbf[idx] = (__bf16)w[(size_t)m * KTOT + f];
}

// Bilinear-gather 8 consecutive taps (f = c*9+k, advancing) for one pixel,
// returning 8 bf16 packed for a single 16B LDS fragment store.
__device__ __forceinline__ v8bf produce_vals(const float* __restrict__ xb,
                                             const int4* s_lin, const float4* s_w,
                                             int prod_n, int c, int k) {
    v8bf vals;
#pragma unroll
    for (int v = 0; v < 8; ++v) {
        int4   lin = s_lin[(k << 6) + prod_n];
        float4 wt  = s_w  [(k << 6) + prod_n];
        const float* pl = xb + (size_t)c * HW_;
        float val = wt.x * pl[lin.x] + wt.y * pl[lin.y]
                  + wt.z * pl[lin.z] + wt.w * pl[lin.w];
        vals[v] = (__bf16)val;
        ++k; if (k == 9) { k = 0; ++c; }
    }
    return vals;
}

// -------------------------------------------------------------------------
// Kernel 2: fused deformable im2col (double-buffered LDS, B-fragment order)
// + software-pipelined WMMA GEMM. Block = 256 threads (8 waves).
// Tile: all 256 output channels x 64 pixels.
// -------------------------------------------------------------------------
__global__ __launch_bounds__(256)
void deform_conv_wmma_kernel(const float* __restrict__ x,
                             const float* __restrict__ offset,
                             const v16bf* __restrict__ wfrag,
                             float* __restrict__ out) {
    __shared__ int4   s_lin[KTAPS * NTILE];     // 4 clamped corner offsets (in-plane)
    __shared__ float4 s_w  [KTAPS * NTILE];     // 4 bilinear weights (validity folded)
    __shared__ __bf16 s_cols[2][4 * 32 * 16];   // double-buffered B fragments, 2x4 KB

    const int tid      = threadIdx.x;
    const int tileBase = blockIdx.x * NTILE;    // global pixel base
    const int b        = tileBase >> 12;        // batch (4096 pixels per image)
    const int remBase  = tileBase & 4095;       // ho*64+wo base

    // ---- per-tile sampling-coordinate precompute (reused by all 256 channels) ----
    for (int e = tid; e < KTAPS * NTILE; e += 256) {
        int k  = e >> 6;            // tap 0..8
        int nl = e & 63;            // pixel in tile
        int rem = remBase + nl;
        int ho = rem >> 6, wo = rem & 63;
        float offy = offset[(size_t)(b * 18 + k * 2 + 0) * HW_ + rem];
        float offx = offset[(size_t)(b * 18 + k * 2 + 1) * HW_ + rem];
        float sy = offy + (float)(ho - 1 + k / 3);
        float sx = offx + (float)(wo - 1 + k % 3);
        float fy0 = floorf(sy), fx0 = floorf(sx);
        float dy = sy - fy0, dx = sx - fx0;
        int y0 = (int)fy0, x0 = (int)fx0;
        int y1 = y0 + 1,  x1 = x0 + 1;
        float vy0 = (y0 >= 0 && y0 < H_) ? 1.f : 0.f;
        float vy1 = (y1 >= 0 && y1 < H_) ? 1.f : 0.f;
        float vx0 = (x0 >= 0 && x0 < W_) ? 1.f : 0.f;
        float vx1 = (x1 >= 0 && x1 < W_) ? 1.f : 0.f;
        int cy0 = min(max(y0, 0), H_ - 1), cy1 = min(max(y1, 0), H_ - 1);
        int cx0 = min(max(x0, 0), W_ - 1), cx1 = min(max(x1, 0), W_ - 1);
        s_lin[e] = make_int4(cy0 * W_ + cx0, cy0 * W_ + cx1,
                             cy1 * W_ + cx0, cy1 * W_ + cx1);
        s_w[e] = make_float4((1.f - dy) * (1.f - dx) * vy0 * vx0,
                             (1.f - dy) * dx         * vy0 * vx1,
                             dy         * (1.f - dx) * vy1 * vx0,
                             dy         * dx         * vy1 * vx1);
    }
    __syncthreads();

    const int wave = tid >> 5;
    const int lane = tid & 31;
    const int m0   = wave * 32;                 // wave owns 32 output channels

    // Producer mapping: thread -> (pixel prod_n, 8 consecutive kk starting prod_kk).
    const int prod_n  = tid >> 2;               // 0..63
    const int prod_kk = (tid & 3) * 8;          // 0,8,16,24
    const int phi = (prod_kk >> 3) & 1;         // lane-half select
    const int pe  = (prod_kk >> 4) << 3;        // element base (0 or 8)
    const int prod_lds = (((prod_n >> 4) * 32) + (prod_n & 15) + phi * 16) * 16 + pe;
    const float* xb = x + (size_t)b * C_ * HW_;

    // incremental (c, k) for f = prod_kk + kt*32 (advance: +3 channels, +5 taps)
    int c0 = prod_kk / 9;
    int k0 = prod_kk - c0 * 9;

    v8f acc[2][4] = {};

    // ---- pipeline prologue: produce chunk 0 into buffer 0 ----
    {
        v8bf vals = produce_vals(xb, s_lin, s_w, prod_n, c0, k0);
        *(v8bf*)&s_cols[0][prod_lds] = vals;
        k0 += 5; c0 += 3; if (k0 >= 9) { k0 -= 9; ++c0; }
    }
    // preload A fragments for kt=0
    v16bf a0 = wfrag[(size_t)(0 * 16 + wave * 2 + 0) * 32 + lane];
    v16bf a1 = wfrag[(size_t)(0 * 16 + wave * 2 + 1) * 32 + lane];

    for (int kt = 0; kt < KSTEPS; ++kt) {
        const int cur = kt & 1;
        __syncthreads();   // buffer[cur] produced by all; buffer[cur^1] fully consumed

        // ---- issue next iteration's VMEM first (hides behind WMMA) ----
        v8bf  nvals;
        v16bf na0, na1;
        if (kt + 1 < KSTEPS) {
            na0 = wfrag[(size_t)((kt + 1) * 16 + wave * 2 + 0) * 32 + lane];
            na1 = wfrag[(size_t)((kt + 1) * 16 + wave * 2 + 1) * 32 + lane];
            nvals = produce_vals(xb, s_lin, s_w, prod_n, c0, k0);
            k0 += 5; c0 += 3; if (k0 >= 9) { k0 -= 9; ++c0; }
        }

        // ---- consume buffer[cur]: 2 M-tiles x 4 N-tiles of WMMA ----
        const v16bf* colsPtr = (const v16bf*)s_cols[cur];
#pragma unroll
        for (int nt = 0; nt < 4; ++nt) {
            v16bf bf = colsPtr[nt * 32 + lane];
            acc[0][nt] = __builtin_amdgcn_wmma_f32_16x16x32_bf16(
                false, a0, false, bf, (short)0, acc[0][nt], false, false);
            acc[1][nt] = __builtin_amdgcn_wmma_f32_16x16x32_bf16(
                false, a1, false, bf, (short)0, acc[1][nt], false, false);
        }

        // ---- stage next chunk into the alternate buffer (safe: barrier above
        //      guarantees iteration kt-1 finished reading buffer[cur^1]) ----
        if (kt + 1 < KSTEPS) {
            *(v8bf*)&s_cols[cur ^ 1][prod_lds] = nvals;
            a0 = na0;
            a1 = na1;
        }
    }

    // ---- writeback: C/D layout -> out[b, m, ho, wo] ----
    const int lhi = (lane >> 4) & 1;
    const int ln  = lane & 15;
#pragma unroll
    for (int mi = 0; mi < 2; ++mi) {
#pragma unroll
        for (int nt = 0; nt < 4; ++nt) {
            int rem = remBase + nt * 16 + ln;
            float* op = out + (size_t)b * (CO_ * HW_) + rem;
#pragma unroll
            for (int r = 0; r < 8; ++r) {
                int m = m0 + mi * 16 + r + lhi * 8;
                op[(size_t)m * HW_] = acc[mi][nt][r];
            }
        }
    }
}

extern "C" void kernel_launch(void* const* d_in, const int* in_sizes, int n_in,
                              void* d_out, int out_size, void* d_ws, size_t ws_size,
                              hipStream_t stream) {
    const float* x      = (const float*)d_in[0];   // [4,256,64,64]
    const float* offset = (const float*)d_in[1];   // [4,18,64,64]
    const float* weight = (const float*)d_in[2];   // [256,256,3,3]
    float* out = (float*)d_out;                    // [4,256,64,64]

    __bf16* wbf = (__bf16*)d_ws;                   // 589824 * 2B = 1.18 MB scratch

    // 72*16*32*16 = 589824 elements -> 2304 blocks of 256
    pack_weights_kernel<<<2304, 256, 0, stream>>>(weight, wbf);
    deform_conv_wmma_kernel<<<PIX / NTILE, 256, 0, stream>>>(
        x, offset, (const v16bf*)wbf, out);
}
